// CausalSelfAttention_52012053954857
// MI455X (gfx1250) — compile-verified, matrix operand order checked by reading, not run on hardware
//
#include <hip/hip_runtime.h>
#include <stdint.h>

// Problem constants (match reference)
#define B_   2
#define T_   2048
#define C_   2048
#define NH   16
#define NKV  4
#define HD   128
#define NREP (NH / NKV)

typedef __attribute__((ext_vector_type(16))) __bf16 v16bf;
typedef __attribute__((ext_vector_type(8)))  float  v8f;

union FragB16 {
    v16bf v;
    uint4 q[2];
};

__device__ __forceinline__ unsigned short f32_to_bf16(float f) {
    unsigned int u = __float_as_uint(f);
    u += 0x7FFFu + ((u >> 16) & 1u);      // round-to-nearest-even
    return (unsigned short)(u >> 16);
}

// A/B bf16 fragment for 16x16x32 WMMA: per lane, two contiguous 16B runs:
// K in {8h..8h+7} and {16+8h..16+8h+7}, h = lane>>4.  p points at run0.
__device__ __forceinline__ FragB16 load_frag(const unsigned short* p) {
    FragB16 f;
    f.q[0] = *(const uint4*)(p);
    f.q[1] = *(const uint4*)(p + 16);
    return f;
}

__device__ __forceinline__ v8f wmma_bf16(FragB16 a, FragB16 b, v8f c) {
    return __builtin_amdgcn_wmma_f32_16x16x32_bf16(
        /*neg_a=*/false, a.v, /*neg_b=*/false, b.v,
        /*c_mod=*/(short)0, c, /*reuse_a=*/false, /*reuse_b=*/false);
}

// ---------------------------------------------------------------- cvt f32->bf16
__global__ void cvt_f32_bf16_kernel(const float* __restrict__ in,
                                    unsigned short* __restrict__ out, int n) {
    int i = blockIdx.x * blockDim.x + threadIdx.x;
    int stride = gridDim.x * blockDim.x;
    for (; i < n; i += stride) out[i] = f32_to_bf16(in[i]);
}

// ---------------------------------------------------------------- GEMM: C[M,N] = A[M,K] * W[N,K]^T
// A, W bf16 (ushort), C f32.  Block = 256 threads = 8 waves (4Mx2N), block tile
// 128x128, wave tile 32x64.  K-step 32, double-buffered LDS staging via
// GLOBAL_LOAD_ASYNC_TO_LDS_B128 (ASYNCcnt), fragments read back as ds_load_b128.
// M,N multiples of 128; K multiple of 32.
__global__ void __launch_bounds__(256)
gemm_bf16_nt_kernel(const unsigned short* __restrict__ A,
                    const unsigned short* __restrict__ W,
                    float* __restrict__ Cout, int M, int N, int K) {
    __shared__ __align__(16) unsigned short Abuf[2][128 * 32];  // 8KB x2
    __shared__ __align__(16) unsigned short Bbuf[2][128 * 32];  // 8KB x2

    const int tid  = threadIdx.x;
    const int lane = tid & 31;
    const int wave = tid >> 5;
    const int l16  = lane & 15;
    const int half = lane >> 4;
    const int wm   = wave >> 1;                 // 0..3 (M)
    const int wn   = wave & 1;                  // 0..1 (N)
    const int m0 = blockIdx.y * 128;
    const int n0 = blockIdx.x * 128;

    const unsigned short* Ag = A + (size_t)m0 * K;
    const unsigned short* Wg = W + (size_t)n0 * K;

    // 32-bit local offsets of the LDS buffers (low 32 bits of flat address)
    const unsigned abase[2] = { (unsigned)(uintptr_t)&Abuf[0][0],
                                (unsigned)(uintptr_t)&Abuf[1][0] };
    const unsigned bbase[2] = { (unsigned)(uintptr_t)&Bbuf[0][0],
                                (unsigned)(uintptr_t)&Bbuf[1][0] };

    const int KT = K >> 5;

    // Fill one 128x32 A tile + 128x32 B tile: 512 chunks of 16B each per tile,
    // 2 A-chunks + 2 B-chunks per thread, issued as async global->LDS copies.
    auto issue = [&](int buf, int k) {
#pragma unroll
        for (int i = 0; i < 2; ++i) {
            const int c   = tid + 256 * i;      // 0..511
            const int row = c >> 2;             // 0..127
            const int sub = c & 3;              // 16B column chunk
            const unsigned short* gpa = Ag + (size_t)row * K + k + sub * 8;
            const unsigned short* gpb = Wg + (size_t)row * K + k + sub * 8;
            __builtin_prefetch(gpa + 32, 0, 1); // warm L2 one K-step ahead
            const unsigned long long ga = (unsigned long long)(uintptr_t)gpa;
            const unsigned long long gb = (unsigned long long)(uintptr_t)gpb;
            const unsigned la = abase[buf] + row * 64 + sub * 16;
            const unsigned lb = bbase[buf] + row * 64 + sub * 16;
            asm volatile("global_load_async_to_lds_b128 %0, %1, off"
                         :: "v"(la), "v"(ga) : "memory");
            asm volatile("global_load_async_to_lds_b128 %0, %1, off"
                         :: "v"(lb), "v"(gb) : "memory");
        }
    };

    v8f acc[2][4] = {};

    issue(0, 0);
    for (int kt = 0; kt < KT; ++kt) {
        const int buf = kt & 1;
        if (kt + 1 < KT) {
            issue(buf ^ 1, (kt + 1) << 5);
            // our 4 newest async loads (next buffer) may stay in flight
            asm volatile("s_wait_asynccnt 4" ::: "memory");
        } else {
            asm volatile("s_wait_asynccnt 0" ::: "memory");
        }
        __syncthreads();   // all threads' fills for `buf` have landed

        FragB16 af[2], bf[4];
#pragma unroll
        for (int i = 0; i < 2; ++i)
            af[i] = load_frag(&Abuf[buf][(wm * 32 + i * 16 + l16) * 32 + 8 * half]);
#pragma unroll
        for (int j = 0; j < 4; ++j)
            bf[j] = load_frag(&Bbuf[buf][(wn * 64 + j * 16 + l16) * 32 + 8 * half]);
#pragma unroll
        for (int i = 0; i < 2; ++i)
#pragma unroll
            for (int j = 0; j < 4; ++j)
                acc[i][j] = wmma_bf16(af[i], bf[j], acc[i][j]);

        __syncthreads();   // everyone done reading `buf` before it is refilled
    }

#pragma unroll
    for (int i = 0; i < 2; ++i)
#pragma unroll
        for (int j = 0; j < 4; ++j)
#pragma unroll
            for (int r = 0; r < 8; ++r)
                Cout[(size_t)(m0 + wm * 32 + i * 16 + r + 8 * half) * N +
                     (n0 + wn * 64 + j * 16 + l16)] = acc[i][j][r];
}

// ---------------------------------------------------------------- QK RMSNorm + RoPE
// One wave per (b, t, idx); idx<NH -> q head, else kv head.
// Writes qhat [B,NH,T,HD] bf16 and khat [B,NKV,T,HD] bf16.
__global__ void __launch_bounds__(128)
qk_prep_kernel(const float* __restrict__ qf, const float* __restrict__ kf,
               const float* __restrict__ qw, const float* __restrict__ kw,
               unsigned short* __restrict__ qhat, unsigned short* __restrict__ khat) {
    const int gw   = (int)((blockIdx.x * (size_t)blockDim.x + threadIdx.x) >> 5);
    const int lane = threadIdx.x & 31;
    const int TOT  = B_ * T_ * (NH + NKV);
    if (gw >= TOT) return;

    const int idx = gw % (NH + NKV);
    const int t   = (gw / (NH + NKV)) % T_;
    const int b   = gw / ((NH + NKV) * T_);
    const bool isq = idx < NH;

    const float* src = isq ? qf + ((size_t)(b * T_ + t)) * C_ + idx * HD
                           : kf + ((size_t)(b * T_ + t)) * (NKV * HD) + (idx - NH) * HD;
    const float* w = isq ? qw : kw;

    float x0 = src[lane], x1 = src[lane + 32], x2 = src[lane + 64], x3 = src[lane + 96];
    float ss = x0 * x0 + x1 * x1 + x2 * x2 + x3 * x3;
    ss += __shfl_xor(ss, 1);
    ss += __shfl_xor(ss, 2);
    ss += __shfl_xor(ss, 4);
    ss += __shfl_xor(ss, 8);
    ss += __shfl_xor(ss, 16);
    const float inv = rsqrtf(ss * (1.0f / (float)HD) + 1e-6f);

    float n0 = x0 * inv * w[lane];
    float n1 = x1 * inv * w[lane + 32];
    float n2 = x2 * inv * w[lane + 64];
    float n3 = x3 * inv * w[lane + 96];

    // RoPE: theta=500000; invf[i] = 2^(-i*log2(theta)/64), i = j mod 64
    const float K2 = 0.29580575889569023f;     // log2(500000)/64
    const float tf = (float)t;
    const float f0 = tf * exp2f(-(float)lane * K2);
    const float f1 = tf * exp2f(-(float)(lane + 32) * K2);
    const float c0 = cosf(f0), s0 = sinf(f0);
    const float c1 = cosf(f1), s1 = sinf(f1);

    const float o0 = n0 * c0 - n2 * s0;       // j = lane
    const float o1 = n1 * c1 - n3 * s1;       // j = lane+32
    const float o2 = n2 * c0 + n0 * s0;       // j = lane+64
    const float o3 = n3 * c1 + n1 * s1;       // j = lane+96

    unsigned short* dst = isq
        ? qhat + (((size_t)(b * NH + idx)) * T_ + t) * HD
        : khat + (((size_t)(b * NKV + (idx - NH))) * T_ + t) * HD;
    dst[lane]      = f32_to_bf16(o0);
    dst[lane + 32] = f32_to_bf16(o1);
    dst[lane + 64] = f32_to_bf16(o2);
    dst[lane + 96] = f32_to_bf16(o3);
}

// ---------------------------------------------------------------- V transpose+convert
// vT[b,kv,d,t] = bf16( vf[(b*T+t)*512 + kv*128 + d] )
__global__ void v_prep_kernel(const float* __restrict__ vf,
                              unsigned short* __restrict__ vT) {
    int i = blockIdx.x * blockDim.x + threadIdx.x;
    int stride = gridDim.x * blockDim.x;
    const int n = B_ * NKV * HD * T_;
    for (; i < n; i += stride) {
        const int t  = i % T_;
        const int d  = (i / T_) % HD;
        const int kv = (i / (T_ * HD)) % NKV;
        const int b  = i / (T_ * HD * NKV);
        vT[i] = f32_to_bf16(vf[(size_t)(b * T_ + t) * (NKV * HD) + kv * HD + d]);
    }
}

// ---------------------------------------------------------------- Flash attention
// One wave per (b, h, 16-query tile).  Block = 128 threads = 4 waves.
// qhat [B,NH,T,HD], khat [B,NKV,T,HD], vT [B,NKV,HD,T] all bf16.
// y bf16 [B*T, NH*HD] (head-major columns).
__global__ void __launch_bounds__(128)
flash_attn_kernel(const unsigned short* __restrict__ qhat,
                  const unsigned short* __restrict__ khat,
                  const unsigned short* __restrict__ vT,
                  unsigned short* __restrict__ y) {
    __shared__ __align__(16) unsigned short ptile[4][16 * 32];

    const int lane = threadIdx.x & 31;
    const int wv   = threadIdx.x >> 5;
    const int l16  = lane & 15;
    const int half = lane >> 4;

    const int wid = (int)(blockIdx.x * 4 + wv);          // over B*NH*(T/16)
    const int qt = wid & (T_ / 16 - 1);
    const int h  = (wid >> 7) & (NH - 1);
    const int b  = wid >> 11;
    const int kv = h / NREP;
    const int q0 = qt * 16;

    const float NEG_INF = __int_as_float(0xff800000);
    const float SCALE   = 0.08838834764831845f;          // 1/sqrt(128)

    // Resident Q fragments: 4 K-steps over HD=128
    const unsigned short* qbase =
        qhat + (((size_t)(b * NH + h)) * T_ + (q0 + l16)) * HD + 8 * half;
    FragB16 qfrag[4];
#pragma unroll
    for (int s = 0; s < 4; ++s) qfrag[s] = load_frag(qbase + s * 32);

    const unsigned short* kbase = khat + ((size_t)(b * NKV + kv)) * T_ * HD;
    const unsigned short* vbase = vT   + ((size_t)(b * NKV + kv)) * HD * T_;

    v8f o[8];
#pragma unroll
    for (int t = 0; t < 8; ++t) o[t] = (v8f){0.f,0.f,0.f,0.f,0.f,0.f,0.f,0.f};
    float mrow[8], lrow[8];
#pragma unroll
    for (int r = 0; r < 8; ++r) { mrow[r] = NEG_INF; lrow[r] = 0.f; }

    for (int j0 = 0; j0 <= q0 + 15; j0 += 32) {          // 32-key chunks, causal bound
        // ---- S = Q * K^T for this chunk (two 16-key tiles)
        v8f sacc[2];
        sacc[0] = (v8f){0.f,0.f,0.f,0.f,0.f,0.f,0.f,0.f};
        sacc[1] = (v8f){0.f,0.f,0.f,0.f,0.f,0.f,0.f,0.f};
#pragma unroll
        for (int s = 0; s < 4; ++s) {
#pragma unroll
            for (int nt = 0; nt < 2; ++nt) {
                FragB16 kf = load_frag(kbase +
                    (size_t)(j0 + nt * 16 + l16) * HD + s * 32 + 8 * half);
                sacc[nt] = wmma_bf16(qfrag[s], kf, sacc[nt]);
            }
        }

        // ---- online softmax (row = q0 + r + 8*half, col = j0 + nt*16 + l16)
        float sv[2][8], p[2][8];
#pragma unroll
        for (int nt = 0; nt < 2; ++nt) {
            const int key = j0 + nt * 16 + l16;
#pragma unroll
            for (int r = 0; r < 8; ++r) {
                const int grow = q0 + r + 8 * half;
                float v = sacc[nt][r] * SCALE;
                sv[nt][r] = (key <= grow) ? v : NEG_INF;
            }
        }
        float alpha[8];
#pragma unroll
        for (int r = 0; r < 8; ++r) {
            float cm = fmaxf(sv[0][r], sv[1][r]);
            cm = fmaxf(cm, __shfl_xor(cm, 1));
            cm = fmaxf(cm, __shfl_xor(cm, 2));
            cm = fmaxf(cm, __shfl_xor(cm, 4));
            cm = fmaxf(cm, __shfl_xor(cm, 8));
            const float mnew = fmaxf(mrow[r], cm);
            alpha[r] = __expf(mrow[r] - mnew);
            mrow[r] = mnew;
        }
#pragma unroll
        for (int nt = 0; nt < 2; ++nt) {
            const int key = j0 + nt * 16 + l16;
#pragma unroll
            for (int r = 0; r < 8; ++r) {
                const int grow = q0 + r + 8 * half;
                p[nt][r] = (key <= grow) ? __expf(sv[nt][r] - mrow[r]) : 0.f;
            }
        }
#pragma unroll
        for (int r = 0; r < 8; ++r) {
            float rs = p[0][r] + p[1][r];
            rs += __shfl_xor(rs, 1);
            rs += __shfl_xor(rs, 2);
            rs += __shfl_xor(rs, 4);
            rs += __shfl_xor(rs, 8);
            lrow[r] = lrow[r] * alpha[r] + rs;
        }
#pragma unroll
        for (int t = 0; t < 8; ++t)
#pragma unroll
            for (int r = 0; r < 8; ++r) o[t][r] *= alpha[r];

        // ---- stage P (16x32, bf16, row-major) through LDS to reshape C->A layout
#pragma unroll
        for (int nt = 0; nt < 2; ++nt)
#pragma unroll
            for (int r = 0; r < 8; ++r)
                ptile[wv][(r + 8 * half) * 32 + nt * 16 + l16] = f32_to_bf16(p[nt][r]);
        asm volatile("s_wait_dscnt 0" ::: "memory");     // same-wave DS RAW fence

        FragB16 pf;
        {
            const unsigned short* pp = &ptile[wv][l16 * 32 + 8 * half];
            pf.q[0] = *(const uint4*)(pp);
            pf.q[1] = *(const uint4*)(pp + 16);
        }

        // ---- O += P * V  (8 HD tiles of 16 columns)
#pragma unroll
        for (int t = 0; t < 8; ++t) {
            FragB16 vf = load_frag(vbase +
                (size_t)(t * 16 + l16) * T_ + j0 + 8 * half);
            o[t] = wmma_bf16(pf, vf, o[t]);
        }
    }

    // ---- normalize and store y (bf16, [B*T, NH*HD])
#pragma unroll
    for (int t = 0; t < 8; ++t) {
#pragma unroll
        for (int r = 0; r < 8; ++r) {
            const int grow = q0 + r + 8 * half;
            y[(size_t)(b * T_ + grow) * C_ + h * HD + t * 16 + l16] =
                f32_to_bf16(o[t][r] / lrow[r]);
        }
    }
}

// ---------------------------------------------------------------- launch
extern "C" void kernel_launch(void* const* d_in, const int* in_sizes, int n_in,
                              void* d_out, int out_size, void* d_ws, size_t ws_size,
                              hipStream_t stream) {
    (void)in_sizes; (void)n_in; (void)out_size; (void)ws_size;
    const float* x   = (const float*)d_in[0];
    const float* wq  = (const float*)d_in[1];
    const float* wk  = (const float*)d_in[2];
    const float* wv  = (const float*)d_in[3];
    const float* wo  = (const float*)d_in[4];
    const float* qnw = (const float*)d_in[5];
    const float* knw = (const float*)d_in[6];

    char* ws = (char*)d_ws;
    // byte offsets (all 256-aligned)
    unsigned short* x_bf  = (unsigned short*)(ws + 0);            // 16 MB
    unsigned short* wq_bf = (unsigned short*)(ws + 16777216);     //  8 MB
    unsigned short* wk_bf = (unsigned short*)(ws + 25165824);     //  2 MB
    unsigned short* wv_bf = (unsigned short*)(ws + 27262976);     //  2 MB
    unsigned short* wo_bf = (unsigned short*)(ws + 29360128);     //  8 MB
    float*          qf32  = (float*)(ws + 37748736);              // 32 MB
    float*          kf32  = (float*)(ws + 71303168);              //  8 MB
    float*          vf32  = (float*)(ws + 79691776);              //  8 MB
    unsigned short* qhat  = (unsigned short*)(ws + 88080384);     // 16 MB
    unsigned short* khat  = (unsigned short*)(ws + 104857600);    //  4 MB
    unsigned short* vTb   = (unsigned short*)(ws + 109051904);    //  4 MB
    unsigned short* y_bf  = (unsigned short*)(ws + 113246208);    // 16 MB

    const int NX = B_ * T_ * C_;            // 8388608
    const int NWQ = C_ * C_;                // 4194304
    const int NWK = NKV * HD * C_;          // 1048576

    // 1) conversions
    cvt_f32_bf16_kernel<<<8192, 256, 0, stream>>>(x,  x_bf,  NX);
    cvt_f32_bf16_kernel<<<8192, 256, 0, stream>>>(wq, wq_bf, NWQ);
    cvt_f32_bf16_kernel<<<4096, 256, 0, stream>>>(wk, wk_bf, NWK);
    cvt_f32_bf16_kernel<<<4096, 256, 0, stream>>>(wv, wv_bf, NWK);
    cvt_f32_bf16_kernel<<<8192, 256, 0, stream>>>(wo, wo_bf, NWQ);

    // 2) QKV projections (M = B*T = 4096; 128x128 block tiles)
    const int M = B_ * T_;
    gemm_bf16_nt_kernel<<<dim3(C_ / 128, M / 128), 256, 0, stream>>>(
        x_bf, wq_bf, qf32, M, C_, C_);
    gemm_bf16_nt_kernel<<<dim3((NKV * HD) / 128, M / 128), 256, 0, stream>>>(
        x_bf, wk_bf, kf32, M, NKV * HD, C_);
    gemm_bf16_nt_kernel<<<dim3((NKV * HD) / 128, M / 128), 256, 0, stream>>>(
        x_bf, wv_bf, vf32, M, NKV * HD, C_);

    // 3) QK norm + RoPE  (one wave per (b,t,head-slot))
    {
        const int waves = B_ * T_ * (NH + NKV);           // 81920
        qk_prep_kernel<<<(waves * 32 + 127) / 128, 128, 0, stream>>>(
            qf32, kf32, qnw, knw, qhat, khat);
    }

    // 4) V transpose + convert
    v_prep_kernel<<<8192, 256, 0, stream>>>(vf32, vTb);

    // 5) flash attention (one wave per (b,h,16-row tile): 4096 waves)
    {
        const int waves = B_ * NH * (T_ / 16);
        flash_attn_kernel<<<waves / 4, 128, 0, stream>>>(qhat, khat, vTb, y_bf);
    }

    // 6) output projection -> d_out (f32)
    gemm_bf16_nt_kernel<<<dim3(C_ / 128, M / 128), 256, 0, stream>>>(
        y_bf, wo_bf, (float*)d_out, M, C_, C_);
}
